// Attention_90177133347352
// MI455X (gfx1250) — matrix operand order repaired and checked by reading, not compiled, not run on hardware
//
#include <hip/hip_runtime.h>
#include <hip/hip_bf16.h>

typedef _Float16 h16;
typedef __attribute__((ext_vector_type(8)))  _Float16 v8h;
typedef __attribute__((ext_vector_type(16))) _Float16 v16h;
typedef __attribute__((ext_vector_type(8)))  float v8f;
typedef __attribute__((ext_vector_type(4)))  int   v4i;

constexpr int kDIM = 2048;
constexpr int kNH  = 16;
constexpr int kNKV = 4;
constexpr int kHD  = 128;
constexpr int kB   = 4;
constexpr int kS   = 1024;
constexpr int kT   = kB * kS;

__device__ __forceinline__ v16h cat16(v8h a, v8h b) {
  union { v16h v; v8h h[2]; } u;
  u.h[0] = a; u.h[1] = b;
  return u.v;
}

__device__ __forceinline__ v8f zero8() {
  v8f z;
  #pragma unroll
  for (int i = 0; i < 8; ++i) z[i] = 0.f;
  return z;
}

__device__ __forceinline__ v8f wmma_f16(v16h a, v16h b, v8f c) {
  return __builtin_amdgcn_wmma_f32_16x16x32_f16(false, a, false, b, (short)0, c, false, false);
}

// ---------------------------------------------------------------------------
// Async global->LDS 16-byte copy (GLOBAL_LOAD_ASYNC_TO_LDS_B128, ASYNCcnt).
// ---------------------------------------------------------------------------
#if __has_builtin(__builtin_amdgcn_global_load_async_to_lds_b128)
#define HAVE_ASYNC_LDS 1
typedef __attribute__((address_space(1))) v4i gv4i;
typedef __attribute__((address_space(3))) v4i lv4i;
#else
#define HAVE_ASYNC_LDS 0
#endif

__device__ __forceinline__ void cp16_async(h16* lds, const h16* g) {
#if HAVE_ASYNC_LDS
  __builtin_amdgcn_global_load_async_to_lds_b128((gv4i*)g, (lv4i*)lds, 0, 0);
#else
  *(v8h*)lds = *(const v8h*)g;
#endif
}

__device__ __forceinline__ void async_wait0() {
#if HAVE_ASYNC_LDS
#if __has_builtin(__builtin_amdgcn_s_wait_asynccnt)
  __builtin_amdgcn_s_wait_asynccnt(0);
#else
  asm volatile("s_wait_asynccnt 0x0" ::: "memory");
#endif
#endif
}

// ---------------------------------------------------------------------------
// Tensor Data Mover: TENSOR_LOAD_TO_LDS via a 2D D# descriptor (ISA 08 §8).
// This toolchain's builtin takes 6 args:
//   (uint32x4 g0, int32x8 g1, int32x4 g2, int32x4 g3, int32x8 g4, i32 cpol)
// D# group0: [1:0]=count, [63:32]=lds_addr, [120:64]=global_addr, [127:126]=2.
// D# group1: [17:16]=data_size(1=2B), [20]=pad_en, [24:22]=pad_interval,
//            [31:25]=pad_amount, [79:48]=tensor_dim0, [111:80]=tensor_dim1,
//            [127:112]=tile_dim0, [143:128]=tile_dim1, [207:160]=dim0_stride.
// ---------------------------------------------------------------------------
#if __has_builtin(__builtin_amdgcn_tensor_load_to_lds) && \
    __has_builtin(__builtin_amdgcn_s_wait_tensorcnt)
#define HAVE_TDM 1
typedef __attribute__((ext_vector_type(4))) unsigned int v4u;
typedef __attribute__((ext_vector_type(8))) int v8i;

__device__ __forceinline__ void tdm_load_2d(const h16* gsrc, unsigned lds_off,
                                            unsigned tile_d0, unsigned tile_d1,
                                            unsigned stride0,
                                            int pad_interval, int pad_amount) {
  unsigned long long ga = (unsigned long long)(size_t)gsrc;
  v4u g0;
  g0[0] = 1u;                                           // count=1, user mode
  g0[1] = lds_off;                                      // LDS byte address
  g0[2] = (unsigned)(ga & 0xFFFFFFFFu);                 // global_addr[31:0]
  g0[3] = (unsigned)((ga >> 32) & 0x1FFFFFFu) | (2u << 30);  // addr[56:32], type=2
  v8i g1;
  g1[0] = (1 << 16) | (1 << 20) | (pad_interval << 22) | (pad_amount << 25);
  g1[1] = (int)((tile_d0 & 0xFFFFu) << 16);             // tensor_dim0 = tile_d0
  g1[2] = (int)(((tile_d0 >> 16) & 0xFFFFu) | ((tile_d1 & 0xFFFFu) << 16));
  g1[3] = (int)(((tile_d1 >> 16) & 0xFFFFu) | ((tile_d0 & 0xFFFFu) << 16)); // tile_dim0
  g1[4] = (int)(tile_d1 & 0xFFFFu);                     // tile_dim1; tile_dim2=0
  g1[5] = (int)stride0;                                 // tensor_dim0_stride[31:0]
  g1[6] = 0;
  g1[7] = 0;
  v4i z4;
  z4[0] = 0; z4[1] = 0; z4[2] = 0; z4[3] = 0;
  v8i z8;
  #pragma unroll
  for (int i = 0; i < 8; ++i) z8[i] = 0;
  __builtin_amdgcn_tensor_load_to_lds(g0, g1, z4, z4, z8, 0);
}

__device__ __forceinline__ void tensor_wait0() {
  __builtin_amdgcn_s_wait_tensorcnt(0);
}
#else
#define HAVE_TDM 0
__device__ __forceinline__ void tensor_wait0() {}
#endif

// ---------------------------------------------------------------------------
// fp32 -> fp16 conversion (grid-stride)
// ---------------------------------------------------------------------------
__global__ void f32_to_f16(const float* __restrict__ in, h16* __restrict__ out, int n) {
  int i = blockIdx.x * blockDim.x + threadIdx.x;
  int stride = gridDim.x * blockDim.x;
  for (; i < n; i += stride) out[i] = (h16)in[i];
}

// ---------------------------------------------------------------------------
// Tiled transpose + convert: Wt[N][K] (f16) = W[K][N] (f32). K,N mult of 32.
// ---------------------------------------------------------------------------
__global__ __launch_bounds__(256) void conv_transpose(
    const float* __restrict__ W, h16* __restrict__ Wt, int K, int N) {
  __shared__ float tile[32][33];
  const int n0 = blockIdx.x * 32;
  const int k0 = blockIdx.y * 32;
  const int tx = threadIdx.x & 31;
  const int ty = (threadIdx.x >> 5) * 4;
  #pragma unroll
  for (int i = 0; i < 4; ++i)
    tile[ty + i][tx] = W[(size_t)(k0 + ty + i) * N + n0 + tx];
  __syncthreads();
  #pragma unroll
  for (int i = 0; i < 4; ++i)
    Wt[(size_t)(n0 + ty + i) * K + k0 + tx] = (h16)tile[tx][ty + i];
}

// ---------------------------------------------------------------------------
// V transpose (f16): Vt[(b*NKV+g)*HD + d][s] = V[b*S+s][g*HD+d]
// ---------------------------------------------------------------------------
__global__ __launch_bounds__(256) void transpose_v(
    const h16* __restrict__ V, h16* __restrict__ Vt) {
  __shared__ h16 tile[32][33];
  const int s0 = blockIdx.x * 32;
  const int c0 = blockIdx.y * 32;
  const int b  = blockIdx.z;
  const int tx = threadIdx.x & 31;
  const int ty = (threadIdx.x >> 5) * 4;
  #pragma unroll
  for (int i = 0; i < 4; ++i)
    tile[ty + i][tx] = V[(size_t)(b * kS + s0 + ty + i) * (kNKV * kHD) + c0 + tx];
  __syncthreads();
  #pragma unroll
  for (int i = 0; i < 4; ++i)
    Vt[(size_t)(b * (kNKV * kHD) + c0 + ty + i) * kS + s0 + tx] = tile[tx][ty + i];
}

// ---------------------------------------------------------------------------
// RoPE on fp16 Q or K. fc: (T, 64, 2) f32 (cos, sin).
// ---------------------------------------------------------------------------
__global__ void rope_f16(h16* __restrict__ t, const float* __restrict__ fc,
                         int heads, int total) {
  int id = blockIdx.x * blockDim.x + threadIdx.x;
  if (id >= total) return;
  int p    = id & 63;
  int rest = id >> 6;
  int hh   = rest % heads;
  int tok  = rest / heads;
  float cs = fc[tok * 128 + 2 * p];
  float sn = fc[tok * 128 + 2 * p + 1];
  size_t base = (size_t)tok * heads * 128 + (size_t)hh * 128 + 2 * p;
  float a = (float)t[base], b = (float)t[base + 1];
  t[base]     = (h16)(a * cs - b * sn);
  t[base + 1] = (h16)(a * sn + b * cs);
}

// ---------------------------------------------------------------------------
// WMMA GEMM: C[MxN] = A[MxK](f16) * Bt[NxK](f16)^T. 128x128x32 tiles,
// double-buffered async LDS staging, 8 waves x (4x2) 16x16 WMMA subtiles.
// ---------------------------------------------------------------------------
__global__ __launch_bounds__(256) void gemm_h16(
    const h16* __restrict__ A, const h16* __restrict__ Bt,
    float* __restrict__ Cf, h16* __restrict__ Ch,
    int M, int N, int K, int outHalf)
{
  __shared__ __align__(16) h16 sA[2][128 * 40];
  __shared__ __align__(16) h16 sB[2][128 * 40];

  const int tid  = threadIdx.x;
  const int lane = tid & 31;
  const int w    = tid >> 5;
  const int lo   = lane & 15;
  const int hi   = lane >> 4;
  const int wr   = w >> 2;
  const int wc   = w & 3;
  const int bm   = blockIdx.y * 128;
  const int bn   = blockIdx.x * 128;

  auto stage = [&](int buf, int k0) {
    #pragma unroll
    for (int it = 0; it < 2; ++it) {
      int c = it * 256 + tid;
      int row = c >> 2, kc = c & 3;
      cp16_async(&sA[buf][row * 40 + kc * 8], A + (size_t)(bm + row) * K + k0 + kc * 8);
    }
    #pragma unroll
    for (int it = 0; it < 2; ++it) {
      int c = it * 256 + tid;
      int n = c >> 2, kc = c & 3;
      cp16_async(&sB[buf][n * 40 + kc * 8], Bt + (size_t)(bn + n) * K + k0 + kc * 8);
    }
  };

  v8f acc[4][2];
  #pragma unroll
  for (int i = 0; i < 4; ++i)
    #pragma unroll
    for (int j = 0; j < 2; ++j) acc[i][j] = zero8();

  stage(0, 0);
  async_wait0();
  __syncthreads();

  int cur = 0;
  for (int k0 = 0; k0 < K; k0 += 32) {
    if (k0 + 32 < K) stage(cur ^ 1, k0 + 32);

    v16h af[4], bf[2];
    #pragma unroll
    for (int i = 0; i < 4; ++i) {
      const int row = wr * 64 + i * 16 + lo;
      af[i] = cat16(*(const v8h*)&sA[cur][row * 40 + 8 * hi],
                    *(const v8h*)&sA[cur][row * 40 + 16 + 8 * hi]);
    }
    #pragma unroll
    for (int j = 0; j < 2; ++j) {
      const int n = wc * 32 + j * 16 + lo;
      bf[j] = cat16(*(const v8h*)&sB[cur][n * 40 + 8 * hi],
                    *(const v8h*)&sB[cur][n * 40 + 16 + 8 * hi]);
    }
    #pragma unroll
    for (int i = 0; i < 4; ++i)
      #pragma unroll
      for (int j = 0; j < 2; ++j)
        acc[i][j] = wmma_f16(af[i], bf[j], acc[i][j]);

    async_wait0();
    __syncthreads();
    cur ^= 1;
  }

  #pragma unroll
  for (int i = 0; i < 4; ++i)
    #pragma unroll
    for (int j = 0; j < 2; ++j)
      #pragma unroll
      for (int p = 0; p < 8; ++p) {
        size_t row = (size_t)(bm + wr * 64 + i * 16 + p + 8 * hi);
        size_t col = (size_t)(bn + wc * 32 + j * 16 + lo);
        if (outHalf) Ch[row * N + col] = (h16)acc[i][j][p];
        else         Cf[row * N + col] = acc[i][j][p];
      }
}

// ---------------------------------------------------------------------------
// Flash attention (causal, GQA). Grid: (S/128, NH, B). Wave w owns 16 query
// rows. K staged [key][d]; V pre-transposed, staged [d][key]. 64-key chunks,
// double-buffered; staging via TDM descriptors (padded rows generated by the
// D# pad fields) when available, else per-thread async b128 copies.
// ---------------------------------------------------------------------------
__global__ __launch_bounds__(256) void flash_attn(
    const h16* __restrict__ Q, const h16* __restrict__ Kc,
    const h16* __restrict__ Vt, h16* __restrict__ O)
{
  __shared__ __align__(16) h16 sK[2][64 * 136];   // [key][d], 128+8 pad
  __shared__ __align__(16) h16 sV[2][128 * 72];   // [d][key], 64+8 pad
  __shared__ __align__(16) h16 sP[8 * 16 * 32];

  const int tid  = threadIdx.x;
  const int lane = tid & 31;
  const int w    = tid >> 5;
  const int lo   = lane & 15;
  const int hi   = lane >> 4;
  const int qblock = blockIdx.x;
  const int h      = blockIdx.y;
  const int g      = h >> 2;
  const int b      = blockIdx.z;
  const int qb0    = qblock * 128 + w * 16;
  const float scale = 0.08838834764831845f;

  auto stage = [&](int buf, int kc0) {
#if HAVE_TDM
    if (w == 0) {
      // K tile: 128 elems/row (64 DW -> pad_interval=5), 64 rows, stride 512,
      // pad 4 DW (code 3) -> 136-half LDS rows.
      tdm_load_2d(Kc + (size_t)(b * kS + kc0) * (kNKV * kHD) + (size_t)g * kHD,
                  (unsigned)(size_t)&sK[buf][0], 128, 64, kNKV * kHD, 5, 3);
      // V tile: 64 elems/row (32 DW -> pad_interval=4), 128 rows, stride 1024,
      // pad 4 DW -> 72-half LDS rows.
      tdm_load_2d(Vt + (size_t)((b * kNKV + g) * kHD) * kS + kc0,
                  (unsigned)(size_t)&sV[buf][0], 64, 128, kS, 4, 3);
    }
#else
    #pragma unroll
    for (int it = 0; it < 4; ++it) {
      int cc = it * 256 + tid;
      int key = cc >> 4, dc = cc & 15;
      cp16_async(&sK[buf][key * 136 + dc * 8],
                 Kc + (size_t)(b * kS + kc0 + key) * (kNKV * kHD) + (size_t)g * kHD + dc * 8);
    }
    #pragma unroll
    for (int it = 0; it < 4; ++it) {
      int cc = it * 256 + tid;
      int d = cc >> 3, kch = cc & 7;
      cp16_async(&sV[buf][d * 72 + kch * 8],
                 Vt + (size_t)((b * kNKV + g) * kHD + d) * kS + kc0 + kch * 8);
    }
#endif
  };

  auto stage_wait = [&]() {
#if HAVE_TDM
    tensor_wait0();
#else
    async_wait0();
#endif
  };

  v16h qf[4];
  {
    size_t qbase = ((size_t)(b * kS + qb0 + lo)) * (kNH * kHD) + (size_t)h * kHD;
    #pragma unroll
    for (int s = 0; s < 4; ++s)
      qf[s] = cat16(*(const v8h*)(Q + qbase + s * 32 + 8 * hi),
                    *(const v8h*)(Q + qbase + s * 32 + 16 + 8 * hi));
  }

  v8f accO[8];
  #pragma unroll
  for (int dt = 0; dt < 8; ++dt) accO[dt] = zero8();
  float mrow[8], lrow[8];
  #pragma unroll
  for (int p = 0; p < 8; ++p) { mrow[p] = -1e30f; lrow[p] = 0.f; }

  const int nch = 2 * (qblock + 1);
  stage(0, 0);
  stage_wait();
  __syncthreads();

  int cur = 0;
  for (int c = 0; c < nch; ++c) {
    const int kc0 = c * 64;
    if (c + 1 < nch) stage(cur ^ 1, kc0 + 64);

    #pragma unroll
    for (int jt = 0; jt < 2; ++jt) {
      const int kbase = kc0 + jt * 32;
      if (kbase > qb0 + 15) continue;

      v8f sc[2];
      #pragma unroll
      for (int j = 0; j < 2; ++j) {
        v8f a = zero8();
        const int krow = jt * 32 + j * 16 + lo;
        #pragma unroll
        for (int s = 0; s < 4; ++s) {
          v16h bf = cat16(*(const v8h*)&sK[cur][krow * 136 + s * 32 + 8 * hi],
                          *(const v8h*)&sK[cur][krow * 136 + s * 32 + 16 + 8 * hi]);
          a = wmma_f16(qf[s], bf, a);
        }
        #pragma unroll
        for (int p = 0; p < 8; ++p) {
          float sv   = a[p] * scale;
          int   qrow = qb0 + p + 8 * hi;
          int   key  = kbase + j * 16 + lo;
          a[p] = (key > qrow) ? -1e30f : sv;
        }
        sc[j] = a;
      }

      float mnew[8], fex[8], psum[8];
      #pragma unroll
      for (int p = 0; p < 8; ++p) {
        float rm = fmaxf(sc[0][p], sc[1][p]);
        #pragma unroll
        for (int msk = 1; msk < 16; msk <<= 1) rm = fmaxf(rm, __shfl_xor(rm, msk, 32));
        mnew[p] = fmaxf(mrow[p], rm);
        fex[p]  = __expf(mrow[p] - mnew[p]);
      }

      h16* pw = &sP[w * 512];
      #pragma unroll
      for (int p = 0; p < 8; ++p) {
        float e0 = __expf(sc[0][p] - mnew[p]);
        float e1 = __expf(sc[1][p] - mnew[p]);
        pw[(p + 8 * hi) * 32 + lo]      = (h16)e0;
        pw[(p + 8 * hi) * 32 + 16 + lo] = (h16)e1;
        float ls = e0 + e1;
        #pragma unroll
        for (int msk = 1; msk < 16; msk <<= 1) ls += __shfl_xor(ls, msk, 32);
        psum[p] = ls;
      }
      asm volatile("s_wait_dscnt 0x0" ::: "memory");

      v16h pf = cat16(*(const v8h*)&pw[lo * 32 + 8 * hi],
                      *(const v8h*)&pw[lo * 32 + 16 + 8 * hi]);
      #pragma unroll
      for (int dt = 0; dt < 8; ++dt) {
        v8f o = accO[dt];
        #pragma unroll
        for (int p = 0; p < 8; ++p) o[p] *= fex[p];
        v16h vf = cat16(*(const v8h*)&sV[cur][(dt * 16 + lo) * 72 + jt * 32 + 8 * hi],
                        *(const v8h*)&sV[cur][(dt * 16 + lo) * 72 + jt * 32 + 16 + 8 * hi]);
        accO[dt] = wmma_f16(pf, vf, o);
      }
      #pragma unroll
      for (int p = 0; p < 8; ++p) {
        lrow[p] = lrow[p] * fex[p] + psum[p];
        mrow[p] = mnew[p];
      }
    }

    stage_wait();
    __syncthreads();
    cur ^= 1;
  }

  #pragma unroll
  for (int dt = 0; dt < 8; ++dt)
    #pragma unroll
    for (int p = 0; p < 8; ++p) {
      size_t row = (size_t)(b * kS + qb0 + p + 8 * hi);
      O[row * (kNH * kHD) + h * kHD + dt * 16 + lo] = (h16)(accO[dt][p] / lrow[p]);
    }
}

// ---------------------------------------------------------------------------
// Host launcher
// ---------------------------------------------------------------------------
extern "C" void kernel_launch(void* const* d_in, const int* in_sizes, int n_in,
                              void* d_out, int out_size, void* d_ws, size_t ws_size,
                              hipStream_t stream) {
  (void)in_sizes; (void)n_in; (void)out_size; (void)ws_size;
  const float* x  = (const float*)d_in[0];
  const float* fc = (const float*)d_in[1];
  const float* wq = (const float*)d_in[2];
  const float* wk = (const float*)d_in[3];
  const float* wv = (const float*)d_in[4];
  const float* wo = (const float*)d_in[5];
  float* out = (float*)d_out;

  h16* p = (h16*)d_ws;
  h16* x16 = p; p += (size_t)kT * kDIM;
  h16* q16 = p; p += (size_t)kT * kNH * kHD;
  h16* k16 = p; p += (size_t)kT * kNKV * kHD;
  h16* v16 = p; p += (size_t)kT * kNKV * kHD;
  h16* vt  = p; p += (size_t)kT * kNKV * kHD;
  h16* a16 = p; p += (size_t)kT * kNH * kHD;
  h16* wqt = p; p += (size_t)kDIM * kNH * kHD;
  h16* wkt = p; p += (size_t)kDIM * kNKV * kHD;
  h16* wvt = p; p += (size_t)kDIM * kNKV * kHD;
  h16* wot = p; p += (size_t)kNH * kHD * kDIM;

  f32_to_f16<<<1024, 256, 0, stream>>>(x, x16, kT * kDIM);
  conv_transpose<<<dim3((kNH * kHD) / 32, kDIM / 32), 256, 0, stream>>>(wq, wqt, kDIM, kNH * kHD);
  conv_transpose<<<dim3((kNKV * kHD) / 32, kDIM / 32), 256, 0, stream>>>(wk, wkt, kDIM, kNKV * kHD);
  conv_transpose<<<dim3((kNKV * kHD) / 32, kDIM / 32), 256, 0, stream>>>(wv, wvt, kDIM, kNKV * kHD);
  conv_transpose<<<dim3(kDIM / 32, (kNH * kHD) / 32), 256, 0, stream>>>(wo, wot, kNH * kHD, kDIM);

  gemm_h16<<<dim3((kNH * kHD) / 128, kT / 128), 256, 0, stream>>>(
      x16, wqt, nullptr, q16, kT, kNH * kHD, kDIM, 1);
  gemm_h16<<<dim3((kNKV * kHD) / 128, kT / 128), 256, 0, stream>>>(
      x16, wkt, nullptr, k16, kT, kNKV * kHD, kDIM, 1);
  gemm_h16<<<dim3((kNKV * kHD) / 128, kT / 128), 256, 0, stream>>>(
      x16, wvt, nullptr, v16, kT, kNKV * kHD, kDIM, 1);

  rope_f16<<<(kT * kNH * 64 + 255) / 256, 256, 0, stream>>>(q16, fc, kNH, kT * kNH * 64);
  rope_f16<<<(kT * kNKV * 64 + 255) / 256, 256, 0, stream>>>(k16, fc, kNKV, kT * kNKV * 64);
  transpose_v<<<dim3(kS / 32, (kNKV * kHD) / 32, kB), 256, 0, stream>>>(v16, vt);

  flash_attn<<<dim3(kS / 128, kNH, kB), 256, 0, stream>>>(q16, k16, vt, a16);

  gemm_h16<<<dim3(kDIM / 128, kT / 128), 256, 0, stream>>>(
      a16, wot, out, nullptr, kT, kDIM, kNH * kHD, 0);
}
